// ContrastiveLearning_16973710754120
// MI455X (gfx1250) — 16-bit and 8-bit matrix operand order repaired and checked by reading, compiled, not run on hardware
//
#include <hip/hip_runtime.h>
#include <stdint.h>


#define BB    8192
#define DD    512
#define LL    1024
#define INV_T 14.2857142857142857f

typedef __attribute__((ext_vector_type(8)))  _Float16 v8h_t;
typedef __attribute__((ext_vector_type(16))) _Float16 v16h;
typedef __attribute__((ext_vector_type(2)))  int      v2i;
typedef __attribute__((ext_vector_type(4)))  int      v4i;
typedef __attribute__((ext_vector_type(8)))  int      v8i;
typedef __attribute__((ext_vector_type(8)))  float    v8f;

__device__ __forceinline__ v16h cat8h(v8h_t a, v8h_t b) {
  return __builtin_shufflevector(a, b, 0,1,2,3,4,5,6,7,8,9,10,11,12,13,14,15);
}
__device__ __forceinline__ v8i cat4i(v4i a, v4i b) {
  return __builtin_shufflevector(a, b, 0,1,2,3,4,5,6,7);
}
__device__ __forceinline__ v4i cat2i(v2i a, v2i b) {
  return __builtin_shufflevector(a, b, 0,1,2,3);
}

__device__ __forceinline__ v16h load_a_h(const _Float16* rowp, int kb, int hi) {
  v8h_t lo = *(const v8h_t*)(rowp + kb + hi * 8);
  v8h_t hi8 = *(const v8h_t*)(rowp + kb + 16 + hi * 8);
  return cat8h(lo, hi8);
}
__device__ __forceinline__ v16h load_b_h(const _Float16* colp, int kb, int hi) {
  v8h_t lo = *(const v8h_t*)(colp + kb + hi * 8);
  v8h_t hi8 = *(const v8h_t*)(colp + kb + 16 + hi * 8);
  return cat8h(lo, hi8);
}
__device__ __forceinline__ v8i load_a_8(const uint8_t* rowp, int kb, int hi) {
  v2i c0 = *(const v2i*)(rowp + kb + hi * 8 + 0);
  v2i c1 = *(const v2i*)(rowp + kb + hi * 8 + 16);
  v2i c2 = *(const v2i*)(rowp + kb + hi * 8 + 32);
  v2i c3 = *(const v2i*)(rowp + kb + hi * 8 + 48);
  return cat4i(cat2i(c0, c1), cat2i(c2, c3));
}
__device__ __forceinline__ v8i load_b_8(const uint8_t* colp, int kb, int hi) {
  v2i c0 = *(const v2i*)(colp + kb + hi * 8 + 0);
  v2i c1 = *(const v2i*)(colp + kb + hi * 8 + 16);
  v2i c2 = *(const v2i*)(colp + kb + hi * 8 + 32);
  v2i c3 = *(const v2i*)(colp + kb + hi * 8 + 48);
  return cat4i(cat2i(c0, c1), cat2i(c2, c3));
}

__global__ void __launch_bounds__(256) prep_rows(const float* __restrict__ zI,
                                                 const float* __restrict__ noise,
                                                 _Float16* __restrict__ zh,
                                                 float* __restrict__ pos) {
  const int lane = threadIdx.x & 31;
  const int w    = threadIdx.x >> 5;
  const int i    = blockIdx.x * 8 + w;
  const float* zr = zI    + (size_t)i * DD;
  const float* nr = noise + (size_t)i * DD;
  _Float16*    hr = zh    + (size_t)i * DD;

  float szz = 0.f, szn = 0.f, snn = 0.f;
#pragma unroll
  for (int t = 0; t < 16; ++t) {
    int k = t * 32 + lane;
    float z = zr[k], n = nr[k];
    szz += z * z;
    szn += z * n;
    snn += n * n;
    hr[k] = (_Float16)z;
  }
#pragma unroll
  for (int off = 16; off > 0; off >>= 1) {
    szz += __shfl_xor(szz, off, 32);
    szn += __shfl_xor(szn, off, 32);
    snn += __shfl_xor(snn, off, 32);
  }
  if (lane == 0) {
    float s_in = szz + szn;
    float s_aa = szz + 2.f * szn + snn;
    float na = fmaxf(sqrtf(szz), 1e-8f);
    float nb = fmaxf(sqrtf(s_aa), 1e-8f);
    pos[i] = (s_in / (na * nb)) * INV_T;
  }
}

__global__ void __launch_bounds__(256) prep_labels(const float* __restrict__ labels,
                                                   uint8_t* __restrict__ lab, int total) {
  for (int idx = blockIdx.x * blockDim.x + threadIdx.x; idx < total;
       idx += gridDim.x * blockDim.x)
    lab[idx] = (labels[idx] != 0.0f) ? (uint8_t)1 : (uint8_t)0;
}

__global__ void __launch_bounds__(256) sim_kernel(const _Float16* __restrict__ zh,
                                                  const uint8_t* __restrict__ lab,
                                                  float* __restrict__ partial) {
  const int lane = threadIdx.x & 31;
  const int w    = threadIdx.x >> 5;
  const int hi   = lane >> 4;
  const int nl   = lane & 15;
  const int ri   = blockIdx.x * 128 + w * 16;
  const int colBase = blockIdx.y * 1024;

  const _Float16* arow  = zh  + (size_t)(ri + nl) * DD;
  const uint8_t*  arow8 = lab + (size_t)(ri + nl) * LL;

  v16h Ah[16];
  v8i  A8[16];
#pragma unroll
  for (int kc = 0; kc < 16; ++kc) Ah[kc] = load_a_h(arow, kc * 32, hi);
#pragma unroll
  for (int kc = 0; kc < 16; ++kc) A8[kc] = load_a_8(arow8, kc * 64, hi);

  float negacc[8];
#pragma unroll
  for (int r = 0; r < 8; ++r) negacc[r] = 0.f;

  for (int j = 0; j < 64; ++j) {
    const int jcol = colBase + j * 16 + nl;
    const _Float16* brow  = zh  + (size_t)jcol * DD;
    const uint8_t*  brow8 = lab + (size_t)jcol * LL;

    if (j < 63) {
      __builtin_prefetch(brow  + 16 * DD, 0, 1);
      __builtin_prefetch(brow8 + 16 * LL, 0, 1);
    }

    v8f c  = {0.f, 0.f, 0.f, 0.f, 0.f, 0.f, 0.f, 0.f};
    v8i ci = {0, 0, 0, 0, 0, 0, 0, 0};

#pragma unroll
    for (int kc = 0; kc < 16; ++kc) {
      v16h bh = load_b_h(brow, kc * 32, hi);
      c = __builtin_amdgcn_wmma_f32_16x16x32_f16(false, Ah[kc], false, bh,
                                                 (short)0, c, false, false);
    }
#pragma unroll
    for (int kc = 0; kc < 16; ++kc) {
      v8i b8 = load_b_8(brow8, kc * 64, hi);
      ci = __builtin_amdgcn_wmma_i32_16x16x64_iu8(false, A8[kc], false, b8,
                                                  ci, false, false);
    }

#pragma unroll
    for (int r = 0; r < 8; ++r) {
      int rowg = ri + r + hi * 8;
      bool ok = (ci[r] == 0) && (rowg != jcol);
      negacc[r] += ok ? __expf(c[r] * INV_T) : 0.0f;
    }
  }

#pragma unroll
  for (int r = 0; r < 8; ++r) {
    float v = negacc[r];
    v += __shfl_xor(v, 1, 32);
    v += __shfl_xor(v, 2, 32);
    v += __shfl_xor(v, 4, 32);
    v += __shfl_xor(v, 8, 32);
    if (nl == 0) {
      int rowg = ri + r + hi * 8;
      partial[(size_t)blockIdx.y * BB + rowg] = v;
    }
  }
}

__global__ void __launch_bounds__(256) finalize(const float* __restrict__ pos,
                                                const float* __restrict__ partial,
                                                float* __restrict__ out) {
  __shared__ float red[256];
  const int tid = threadIdx.x;
  float acc = 0.f;
  for (int i = tid; i < BB; i += 256) {
    float neg = 0.f;
#pragma unroll
    for (int cc = 0; cc < 8; ++cc) neg += partial[(size_t)cc * BB + i];
    float p = pos[i];
    acc += logf(expf(p) + neg) - p;
  }
  red[tid] = acc;
  __syncthreads();
  for (int s = 128; s > 0; s >>= 1) {
    if (tid < s) red[tid] += red[tid + s];
    __syncthreads();
  }
  if (tid == 0) out[0] = red[0] / (float)BB;
}

extern "C" void kernel_launch(void* const* d_in, const int* in_sizes, int n_in,
                              void* d_out, int out_size, void* d_ws, size_t ws_size,
                              hipStream_t stream) {
  const float* zI     = (const float*)d_in[0];
  const float* labels = (const float*)d_in[2];
  const float* noise  = (const float*)d_in[3];
  float* out = (float*)d_out;

  char* ws = (char*)d_ws;
  _Float16* zh      = (_Float16*)(ws);
  uint8_t*  lab     = (uint8_t*)(ws + 8388608);
  float*    pos     = (float*)(ws + 16777216);
  float*    partial = (float*)(ws + 16809984);

  prep_rows<<<BB / 8, 256, 0, stream>>>(zI, noise, zh, pos);
  prep_labels<<<4096, 256, 0, stream>>>(labels, lab, BB * LL);

  dim3 grid(BB / 128, BB / 1024);
  sim_kernel<<<grid, 256, 0, stream>>>(zh, lab, partial);

  finalize<<<1, 256, 0, stream>>>(pos, partial, out);
}
